// Model_386547056879
// MI455X (gfx1250) — compile-verified, matrix operand order checked by reading, not run on hardware
//
#include <hip/hip_runtime.h>

typedef __attribute__((ext_vector_type(16))) _Float16 v16h;
typedef __attribute__((ext_vector_type(8)))  _Float16 v8h;
typedef __attribute__((ext_vector_type(8)))  float    v8f;
typedef __attribute__((ext_vector_type(4)))  float    f4;
typedef __attribute__((__vector_size__(16))) int      v4i;
typedef __attribute__((address_space(1))) v4i*        as1_v4i;
typedef __attribute__((address_space(3))) v4i*        as3_v4i;

#define NN   8192
#define NIN  64
#define NH   128
#define NEG_SLOPE 0.2f

// LDS tile row stride in halves (multiple of 8 -> every 8-half chunk is 16B aligned)
#define TSTRIDE 40

#if defined(__has_builtin)
#if __has_builtin(__builtin_amdgcn_global_load_async_to_lds_b128)
#define HAVE_ASYNC_LDS 1
#endif
#endif
#ifndef HAVE_ASYNC_LDS
#define HAVE_ASYNC_LDS 0
#endif

#if defined(__has_builtin)
#if __has_builtin(__builtin_amdgcn_s_wait_asynccnt)
#define WAIT_ASYNC() __builtin_amdgcn_s_wait_asynccnt(0)
#else
#define WAIT_ASYNC() asm volatile("s_wait_asynccnt 0x0" ::: "memory")
#endif
#else
#define WAIT_ASYNC() asm volatile("s_wait_asynccnt 0x0" ::: "memory")
#endif

#define COMBINE16(lo, hi) \
  __builtin_shufflevector((lo), (hi), 0, 1, 2, 3, 4, 5, 6, 7, 8, 9, 10, 11, 12, 13, 14, 15)

// ---------------------------------------------------------------- init
__global__ void k_init(float* __restrict__ emb32, float* __restrict__ out) {
  int i = blockIdx.x * 256 + threadIdx.x;
  if (i < NN * NIN) emb32[i] = 0.f;
  if (i == 0) out[0] = 0.f;
}

// ---------------------------------------------------------------- per-row MLPs
__global__ void k_features(const float* __restrict__ seq1,
                           const float* __restrict__ W_stru, const float* __restrict__ b_stru,
                           const float* __restrict__ W_gat,
                           const float* __restrict__ att_src, const float* __restrict__ att_dst,
                           const float* __restrict__ W_a1, const float* __restrict__ b_a1,
                           const float* __restrict__ W_a2, const float* __restrict__ b_a2,
                           _Float16* __restrict__ hp16T,   // [NIN][NN] (transposed)
                           float* __restrict__ a_s, float* __restrict__ a_d,
                           float* __restrict__ attr_err) {
  __shared__ float srow[NIN];
  __shared__ float h[NH];
  __shared__ float xa[NH];
  __shared__ float reds[NIN], redd[NIN], redx[NIN];
  const int r = blockIdx.x;
  const int t = threadIdx.x;
  if (t < NIN) srow[t] = seq1[r * NIN + t];
  __syncthreads();
  float acc1 = 0.f, acc2 = 0.f;
#pragma unroll 8
  for (int k = 0; k < NIN; ++k) {
    float s = srow[k];
    acc1 += s * W_stru[t * NIN + k];
    acc2 += s * W_a1[t * NIN + k];
  }
  h[t]  = fmaxf(acc1 + b_stru[t], 0.f);
  xa[t] = fmaxf(acc2 + b_a1[t], 0.f);
  __syncthreads();
  if (t < NIN) {
    float hp = 0.f, x2 = 0.f;
#pragma unroll 8
    for (int k = 0; k < NH; ++k) {
      hp += h[k]  * W_gat[t * NH + k];
      x2 += xa[k] * W_a2[t * NH + k];
    }
    x2 += b_a2[t];
    hp16T[(size_t)t * NN + r] = (_Float16)hp;   // transposed: [d][row]
    reds[t] = hp * att_src[t];
    redd[t] = hp * att_dst[t];
    float d = srow[t] - x2;
    redx[t] = d * d;
  }
  __syncthreads();
  if (t == 0) { float s = 0.f; for (int k = 0; k < NIN; ++k) s += reds[k]; a_s[r] = s; }
  if (t == 1) { float s = 0.f; for (int k = 0; k < NIN; ++k) s += redd[k]; a_d[r] = s; }
  if (t == 2) { float s = 0.f; for (int k = 0; k < NIN; ++k) s += redx[k]; attr_err[r] = sqrtf(s); }
}

// ---------------------------------------------------------------- column-wise online (max, sum-exp)
__global__ void k_colstat(const float* __restrict__ adj,
                          const float* __restrict__ a_s, const float* __restrict__ a_d,
                          float* __restrict__ mpart, float* __restrict__ lpart) {
  const int j   = blockIdx.x * 256 + threadIdx.x;
  const int seg = blockIdx.y;
  const float ad = a_d[j];
  float m = -3.0e38f, l = 0.f;
  const int i_begin = seg * (NN / 8);
  const int i_end   = i_begin + (NN / 8);
  for (int i = i_begin; i < i_end; ++i) {
    float v = adj[(size_t)i * NN + j];
    if (v > 0.f || i == j) {                 // GATConv adds self loops
      float x = a_s[i] + ad;
      x = (x > 0.f) ? x : NEG_SLOPE * x;     // leaky_relu
      float nm = fmaxf(m, x);
      l = l * __expf(m - nm) + __expf(x - nm);
      m = nm;
    }
  }
  mpart[seg * NN + j] = m;
  lpart[seg * NN + j] = l;
}

__global__ void k_colmerge(const float* __restrict__ mpart, const float* __restrict__ lpart,
                           float* __restrict__ mcol, float* __restrict__ rlcol) {
  const int j = blockIdx.x * 256 + threadIdx.x;
  float m = -3.0e38f;
#pragma unroll
  for (int s = 0; s < 8; ++s) m = fmaxf(m, mpart[s * NN + j]);
  float l = 0.f;
#pragma unroll
  for (int s = 0; s < 8; ++s) l += lpart[s * NN + j] * __expf(mpart[s * NN + j] - m);
  mcol[j]  = m;
  rlcol[j] = 1.0f / l;                        // l >= 1 (self loop)
}

// ---------------------------------------------------------------- emb = attn^T @ hp  (WMMA)
__global__ void k_emb(const float* __restrict__ adj,
                      const float* __restrict__ a_s, const float* __restrict__ a_d,
                      const float* __restrict__ mcol, const float* __restrict__ rlcol,
                      const _Float16* __restrict__ hp16T,   // [NIN][NN]
                      float* __restrict__ emb32) {
  __shared__ __align__(16) _Float16 PT[128][TSTRIDE];  // [j_local][i_local] (K-major)
  __shared__ __align__(16) _Float16 HT[NIN][TSTRIDE];  // [d][i_local]       (K-major)
  __shared__ float adv[128], mj[128], rlj[128];
  const int t    = threadIdx.x;
  const int wave = t >> 5;
  const int lane = t & 31;
  const int hi   = lane >> 4;
  const int ln   = lane & 15;
  const int j0   = blockIdx.x * 128;
  const int seg  = blockIdx.y;
  if (t < 128) { adv[t] = a_d[j0 + t]; mj[t] = mcol[j0 + t]; rlj[t] = rlcol[j0 + t]; }
  __syncthreads();
  v8f c0 = {}, c1 = {}, c2 = {}, c3 = {};
  const int i_begin = seg * (NN / 8);
  const int i_end   = i_begin + NN / 8;
  for (int i0 = i_begin; i0 < i_end; i0 += 32) {
    // ---- stage hp tile [64 d][32 i] into HT (K-major): one b128 per thread
    {
      const int d   = t >> 2;           // 0..63
      const int s16 = (t & 3) * 8;      // which 8-half chunk of the 32 K values
      const _Float16* gsrc = hp16T + (size_t)d * NN + i0 + s16;
      _Float16* ldst = &HT[d][s16];
#if HAVE_ASYNC_LDS
      __builtin_amdgcn_global_load_async_to_lds_b128(
          (as1_v4i)(uintptr_t)gsrc, (as3_v4i)(uintptr_t)ldst, 0, 0);
#else
      *(v8h*)ldst = *(const v8h*)gsrc;
#endif
    }
    // ---- prefetch next adj tile
    if (i0 + 32 < i_end)
      __builtin_prefetch((const void*)(adj + (size_t)(i0 + 32) * NN + j0 + (t & 127)), 0, 1);
    // ---- build softmax-probability tile, stored transposed PT[j][i]
#pragma unroll
    for (int q = 0; q < 4; ++q) {
      int e  = q * 256 + t;             // 1024 float4 = 32 x 128
      int k  = e >> 5;                  // i_local
      int j4 = (e & 31) * 4;            // j_local base
      int gi = i0 + k;
      f4 v = *(const f4*)(adj + (size_t)gi * NN + j0 + j4);
      float as = a_s[gi];
#pragma unroll
      for (int u = 0; u < 4; ++u) {
        int j = j4 + u;
        float pv = 0.f;
        if (v[u] > 0.f || gi == (j0 + j)) {
          float x = as + adv[j];
          x = (x > 0.f) ? x : NEG_SLOPE * x;
          pv = __expf(x - mj[j]) * rlj[j];
        }
        PT[j][k] = (_Float16)pv;
      }
    }
#if HAVE_ASYNC_LDS
    WAIT_ASYNC();
#endif
    __syncthreads();
    // ---- fragments: two 16B LDS loads each (K runs [hi*8,+7] and [16+hi*8,+7])
    const int jcol = wave * 16 + ln;
    v8h alo = *(const v8h*)&PT[jcol][hi * 8];
    v8h ahi = *(const v8h*)&PT[jcol][16 + hi * 8];
    v16h a = COMBINE16(alo, ahi);
    {
      v8h blo = *(const v8h*)&HT[0 + ln][hi * 8];
      v8h bhi = *(const v8h*)&HT[0 + ln][16 + hi * 8];
      v16h b = COMBINE16(blo, bhi);
      c0 = __builtin_amdgcn_wmma_f32_16x16x32_f16(false, a, false, b, (short)0, c0, false, false);
    }
    {
      v8h blo = *(const v8h*)&HT[16 + ln][hi * 8];
      v8h bhi = *(const v8h*)&HT[16 + ln][16 + hi * 8];
      v16h b = COMBINE16(blo, bhi);
      c1 = __builtin_amdgcn_wmma_f32_16x16x32_f16(false, a, false, b, (short)0, c1, false, false);
    }
    {
      v8h blo = *(const v8h*)&HT[32 + ln][hi * 8];
      v8h bhi = *(const v8h*)&HT[32 + ln][16 + hi * 8];
      v16h b = COMBINE16(blo, bhi);
      c2 = __builtin_amdgcn_wmma_f32_16x16x32_f16(false, a, false, b, (short)0, c2, false, false);
    }
    {
      v8h blo = *(const v8h*)&HT[48 + ln][hi * 8];
      v8h bhi = *(const v8h*)&HT[48 + ln][16 + hi * 8];
      v16h b = COMBINE16(blo, bhi);
      c3 = __builtin_amdgcn_wmma_f32_16x16x32_f16(false, a, false, b, (short)0, c3, false, false);
    }
    __syncthreads();
  }
#pragma unroll
  for (int vv = 0; vv < 8; ++vv) {
    int j = j0 + wave * 16 + vv + hi * 8;    // C layout: M = vgpr + hi*8, N = ln
    atomicAdd(&emb32[j * NIN +  0 + ln], c0[vv]);
    atomicAdd(&emb32[j * NIN + 16 + ln], c1[vv]);
    atomicAdd(&emb32[j * NIN + 32 + ln], c2[vv]);
    atomicAdd(&emb32[j * NIN + 48 + ln], c3[vv]);
  }
}

__global__ void k_embfin(const float* __restrict__ emb32, const float* __restrict__ b_gat,
                         _Float16* __restrict__ emb16) {
  int i = blockIdx.x * 256 + threadIdx.x;
  if (i < NN * NIN) emb16[i] = (_Float16)(emb32[i] + b_gat[i & 63]);
}

// ---------------------------------------------------------------- gathered rows of sigmoid(emb @ emb^T), fused scoring
__global__ void k_score(const _Float16* __restrict__ emb16,   // [NN][NIN] row-major
                        const float* __restrict__ adj,
                        const float* __restrict__ attr_err,
                        const int* __restrict__ idx_train, const int* __restrict__ idx_test,
                        float* __restrict__ out) {
  __shared__ int   rows[16];
  __shared__ float accum[16];
  const int t    = threadIdx.x;
  const int wave = t >> 5;
  const int lane = t & 31;
  const int hi   = lane >> 4;
  const int ln   = lane & 15;
  if (t < 16) {
    int g = blockIdx.x * 16 + t;
    rows[t]  = (g < 4096) ? idx_train[g] : idx_test[g - 4096];
    accum[t] = 0.f;
  }
  __syncthreads();
  // A fragments: K contiguous in emb16 rows -> pure b128 loads
  const v8h* arow = (const v8h*)(emb16 + (size_t)rows[ln] * NIN);
  v16h a0 = COMBINE16(arow[hi], arow[2 + hi]);
  v16h a1 = COMBINE16(arow[4 + hi], arow[6 + hi]);
  float ps[8];
#pragma unroll
  for (int vv = 0; vv < 8; ++vv) ps[vv] = 0.f;
  for (int cb = wave * 16; cb < NN; cb += 128) {
    const int ccol = cb + ln;
    const v8h* brow = (const v8h*)(emb16 + (size_t)ccol * NIN);
    v16h b0 = COMBINE16(brow[hi], brow[2 + hi]);
    v16h b1 = COMBINE16(brow[4 + hi], brow[6 + hi]);
    v8f c = {};
    c = __builtin_amdgcn_wmma_f32_16x16x32_f16(false, a0, false, b0, (short)0, c, false, false);
    c = __builtin_amdgcn_wmma_f32_16x16x32_f16(false, a1, false, b1, (short)0, c, false, false);
#pragma unroll
    for (int vv = 0; vv < 8; ++vv) {
      int M = vv + hi * 8;
      float sig = 1.0f / (1.0f + __expf(-c[vv]));
      float av  = adj[(size_t)rows[M] * NN + ccol];
      float d   = av - sig;
      ps[vv] += d * d;
    }
  }
#pragma unroll
  for (int vv = 0; vv < 8; ++vv) atomicAdd(&accum[vv + hi * 8], ps[vv]);
  __syncthreads();
  if (t < 16) {
    int g = blockIdx.x * 16 + t;
    float stru  = sqrtf(accum[t]);
    float score = 0.5f * attr_err[rows[t]] + 0.5f * stru;
    if (g < 4096) atomicAdd(&out[0], score * (1.0f / 4096.0f));
    else          out[1 + (g - 4096)] = score;
  }
}

// ---------------------------------------------------------------- launch
extern "C" void kernel_launch(void* const* d_in, const int* in_sizes, int n_in,
                              void* d_out, int out_size, void* d_ws, size_t ws_size,
                              hipStream_t stream) {
  const float* seq1    = (const float*)d_in[0];
  const float* adj     = (const float*)d_in[1];
  const int*   idx_tr  = (const int*)d_in[2];
  const int*   idx_te  = (const int*)d_in[3];
  const float* W_stru  = (const float*)d_in[4];
  const float* b_stru  = (const float*)d_in[5];
  const float* W_gat   = (const float*)d_in[6];
  const float* att_src = (const float*)d_in[7];
  const float* att_dst = (const float*)d_in[8];
  const float* b_gat   = (const float*)d_in[9];
  const float* W_a1    = (const float*)d_in[10];
  const float* b_a1    = (const float*)d_in[11];
  const float* W_a2    = (const float*)d_in[12];
  const float* b_a2    = (const float*)d_in[13];
  float* out = (float*)d_out;

  char* ws = (char*)d_ws;
  size_t off = 0;
  auto alloc = [&](size_t bytes) -> void* {
    void* p = ws + off;
    off += (bytes + 255) & ~(size_t)255;
    return p;
  };
  _Float16* hp16T = (_Float16*)alloc((size_t)NN * NIN * 2);
  float*    emb32 = (float*)   alloc((size_t)NN * NIN * 4);
  _Float16* emb16 = (_Float16*)alloc((size_t)NN * NIN * 2);
  float*    a_s   = (float*)   alloc((size_t)NN * 4);
  float*    a_d   = (float*)   alloc((size_t)NN * 4);
  float*    attr  = (float*)   alloc((size_t)NN * 4);
  float*    mpart = (float*)   alloc((size_t)8 * NN * 4);
  float*    lpart = (float*)   alloc((size_t)8 * NN * 4);
  float*    mcol  = (float*)   alloc((size_t)NN * 4);
  float*    rlcol = (float*)   alloc((size_t)NN * 4);

  hipLaunchKernelGGL(k_init, dim3((NN * NIN + 255) / 256), dim3(256), 0, stream, emb32, out);
  hipLaunchKernelGGL(k_features, dim3(NN), dim3(128), 0, stream,
                     seq1, W_stru, b_stru, W_gat, att_src, att_dst,
                     W_a1, b_a1, W_a2, b_a2, hp16T, a_s, a_d, attr);
  hipLaunchKernelGGL(k_colstat, dim3(NN / 256, 8), dim3(256), 0, stream, adj, a_s, a_d, mpart, lpart);
  hipLaunchKernelGGL(k_colmerge, dim3(NN / 256), dim3(256), 0, stream, mpart, lpart, mcol, rlcol);
  hipLaunchKernelGGL(k_emb, dim3(NN / 128, 8), dim3(256), 0, stream,
                     adj, a_s, a_d, mcol, rlcol, hp16T, emb32);
  hipLaunchKernelGGL(k_embfin, dim3((NN * NIN + 255) / 256), dim3(256), 0, stream, emb32, b_gat, emb16);
  hipLaunchKernelGGL(k_score, dim3(NN / 16), dim3(256), 0, stream,
                     emb16, adj, attr, idx_tr, idx_te, out);
}
